// Audio2Landmark_88373247083147
// MI455X (gfx1250) — matrix operand order repaired and checked by reading, not compiled, not run on hardware
//
#include <hip/hip_runtime.h>
#include <hip/hip_bf16.h>
#include <math.h>

// ---------------------------------------------------------------------------
// FaceFormer-style autoregressive decoder for MI455X (gfx1250, wave32, WMMA).
//
//  * All big GEMMs: bf16 WMMA (v_wmma_f32_16x16x32_bf16), f32 accumulate.
//    Weights converted fp32->bf16 once per launch (~30MB, L2-resident).
//  * Activations are produced in bf16 by their producer kernels (PE-add, LN,
//    attention, FF1) so every per-step GEMM uses the pure bf16 A path:
//    inner loop = 4x global_load_b128 + 1 WMMA, zero converts.
//  * B=16 == WMMA M tile: each wave owns one 16x16 N tile, K loop step 32.
//  * Attention / softmax / layernorm stay fp32 VALU (tiny FLOPs, accuracy).
//  * N=1404/K=1404 padded to 1408 so every WMMA K loop is a multiple of 32.
//  * Loss: deterministic per-(step,block) partials + final reduce.
// ---------------------------------------------------------------------------

typedef __bf16 bf16;
typedef __attribute__((ext_vector_type(16))) __bf16 v16bf;
typedef __attribute__((ext_vector_type(8)))  __bf16 v8bf;
typedef __attribute__((ext_vector_type(8)))  float  v8f;

#define B_      16
#define S_      1200
#define T_      600
#define D_      1024
#define V_      1404
#define VP_     1408
#define FF_     2048
#define H_      4
#define DH_     256
#define PERIOD_ 25
#define VR_BLOCKS 22   // 88 N-tiles (1404->1408) / 4 waves

// ---------------- WMMA fragment loaders (wave32 layouts, ISA 7.12.2) -------
// A (16x32 bf16): lane l holds row m=l&15; K chunks {klo..klo+7, klo+16..klo+23},
// klo = (l>=16)?8:0.
__device__ __forceinline__ v16bf load_a_bf16(const bf16* __restrict__ row, int kbase) {
  v8bf lo = *(const v8bf*)(row + kbase);
  v8bf hi = *(const v8bf*)(row + kbase + 16);
  v16bf r;
#pragma unroll
  for (int j = 0; j < 8; ++j) { r[j] = lo[j]; r[j + 8] = hi[j]; }
  return r;
}
__device__ __forceinline__ v16bf load_a_f32(const float* __restrict__ row, int kbase) {
  float4 a0 = *(const float4*)(row + kbase);
  float4 a1 = *(const float4*)(row + kbase + 4);
  float4 b0 = *(const float4*)(row + kbase + 16);
  float4 b1 = *(const float4*)(row + kbase + 20);
  v16bf r;
  r[0] = (bf16)a0.x; r[1] = (bf16)a0.y; r[2]  = (bf16)a0.z; r[3]  = (bf16)a0.w;
  r[4] = (bf16)a1.x; r[5] = (bf16)a1.y; r[6]  = (bf16)a1.z; r[7]  = (bf16)a1.w;
  r[8] = (bf16)b0.x; r[9] = (bf16)b0.y; r[10] = (bf16)b0.z; r[11] = (bf16)b0.w;
  r[12]= (bf16)b1.x; r[13]= (bf16)b1.y; r[14] = (bf16)b1.z; r[15] = (bf16)b1.w;
  return r;
}
// B (32x16 bf16): lane l holds col n=l&15; K chunks {klo..klo+7, klo+16..klo+23}, klo = (l>=16)?8:0.
__device__ __forceinline__ v16bf load_b_bf16(const bf16* __restrict__ row, int kbase) {
  v8bf lo = *(const v8bf*)(row + kbase);
  v8bf hi = *(const v8bf*)(row + kbase + 16);
  v16bf r;
#pragma unroll
  for (int j = 0; j < 8; ++j) { r[j] = lo[j]; r[j + 8] = hi[j]; }
  return r;
}

// ---------------- Generic WMMA GEMM:  out(MxN) = A(MxK) @ W(NxK)^T ----------
// block = 128 threads (4 waves); wave w -> N tile blockIdx.x*4+w;
// blockIdx.y -> M tile (row0 = 16*blockIdx.y). lda == K (row-major, padded).
template<bool ABF16, bool RELU, bool OUTBF16>
__global__ __launch_bounds__(128) void gemm_wmma_kernel(
    const void* __restrict__ Av, const bf16* __restrict__ W,
    const float* __restrict__ bias, const float* __restrict__ residual,
    void* __restrict__ outv, int N, int K)
{
  const int lane  = threadIdx.x & 31;
  const int wave  = threadIdx.x >> 5;
  const int tn    = blockIdx.x * 4 + wave;
  const int n     = tn * 16 + (lane & 15);
  const int nclmp = (n < N) ? n : (N - 1);
  const int row0  = blockIdx.y * 16;
  const int m     = lane & 15;
  const int kaoff = (lane >> 4) << 3;   // 0 / 8
  const int kboff = (lane >> 4) << 3;   // 0 / 8

  const bf16*  wrow   = W + (size_t)nclmp * K;
  const bf16*  arow_b = (const bf16*)Av + (size_t)(row0 + m) * K;
  const float* arow_f = (const float*)Av + (size_t)(row0 + m) * K;

  v8f acc = {};
#pragma unroll 2
  for (int kk = 0; kk < K; kk += 32) {
    v16bf afrag;
    if constexpr (ABF16) afrag = load_a_bf16(arow_b, kk + kaoff);
    else                 afrag = load_a_f32 (arow_f, kk + kaoff);
    v16bf bfrag = load_b_bf16(wrow, kk + kboff);
    if (kk + 256 < K) __builtin_prefetch(wrow + kk + 256, 0, 1);
    acc = __builtin_amdgcn_wmma_f32_16x16x32_bf16(
        false, afrag, false, bfrag, (short)0, acc, false, false);
  }
  if (n >= N) return;
  const float bv     = bias ? bias[n] : 0.0f;
  const int   mshift = (lane & 16) ? 8 : 0;
#pragma unroll
  for (int r = 0; r < 8; ++r) {
    const int mr = row0 + r + mshift;
    float v = acc[r] + bv;
    if (residual) v += residual[(size_t)mr * N + n];
    if (RELU) v = fmaxf(v, 0.0f);
    if constexpr (OUTBF16) ((bf16*)outv)[(size_t)mr * N + n] = (bf16)v;
    else                   ((float*)outv)[(size_t)mr * N + n] = v;
  }
}

// ---------------- vertice_map_r GEMM + loss epilogue -----------------------
// out_v(16x1404) = x3 @ W_vr^T + b_vr; writes bf16 padded copy (16x1408) and
// per-block sum of (out_v - vertice)^2. A (x3) already bf16.
__global__ __launch_bounds__(128) void vr_loss_kernel(
    const bf16* __restrict__ x3b, const bf16* __restrict__ Wvr,
    const float* __restrict__ bvr, const float* __restrict__ vertice,
    bf16* __restrict__ outv_pad, float* __restrict__ loss_partial, int step)
{
  __shared__ float red[4];
  const int lane  = threadIdx.x & 31;
  const int wave  = threadIdx.x >> 5;
  const int tn    = blockIdx.x * 4 + wave;
  const int n     = tn * 16 + (lane & 15);
  const int nclmp = (n < V_) ? n : (V_ - 1);
  const int m     = lane & 15;
  const int kaoff = (lane >> 4) << 3;
  const int kboff = (lane >> 4) << 3;
  const bf16* wrow = Wvr + (size_t)nclmp * D_;
  const bf16* arow = x3b + (size_t)m * D_;

  v8f acc = {};
#pragma unroll 2
  for (int kk = 0; kk < D_; kk += 32) {
    v16bf afrag = load_a_bf16(arow, kk + kaoff);
    v16bf bfrag = load_b_bf16(wrow, kk + kboff);
    acc = __builtin_amdgcn_wmma_f32_16x16x32_bf16(
        false, afrag, false, bfrag, (short)0, acc, false, false);
  }
  const int  mshift = (lane & 16) ? 8 : 0;
  const bool valid  = (n < V_);
  const float bv    = valid ? bvr[n] : 0.0f;
  float ls = 0.0f;
#pragma unroll
  for (int r = 0; r < 8; ++r) {
    const int b = r + mshift;                  // batch row
    float y = valid ? (acc[r] + bv) : 0.0f;
    outv_pad[(size_t)b * VP_ + n] = (bf16)y;   // zero-padded cols 1404..1407
    if (valid) {
      float vv = vertice[((size_t)b * T_ + step) * V_ + n];
      float d = y - vv;
      ls += d * d;
    }
  }
  for (int o = 16; o > 0; o >>= 1) ls += __shfl_down(ls, o, 32);
  if (lane == 0) red[wave] = ls;
  __syncthreads();
  if (threadIdx.x == 0)
    loss_partial[step * VR_BLOCKS + blockIdx.x] = red[0] + red[1] + red[2] + red[3];
}

// ---------------- block reductions (256 threads = 8 waves) -----------------
__device__ __forceinline__ float block_sum256(float v, float* red) {
  for (int o = 16; o > 0; o >>= 1) v += __shfl_down(v, o, 32);
  const int lane = threadIdx.x & 31, w = threadIdx.x >> 5;
  __syncthreads();
  if (lane == 0) red[w] = v;
  __syncthreads();
  if (threadIdx.x == 0) {
    float s = 0.0f;
    for (int i = 0; i < 8; ++i) s += red[i];
    red[0] = s;
  }
  __syncthreads();
  return red[0];
}
__device__ __forceinline__ float block_max256(float v, float* red) {
  for (int o = 16; o > 0; o >>= 1) v = fmaxf(v, __shfl_down(v, o, 32));
  const int lane = threadIdx.x & 31, w = threadIdx.x >> 5;
  __syncthreads();
  if (lane == 0) red[w] = v;
  __syncthreads();
  if (threadIdx.x == 0) {
    float s = -1e30f;
    for (int i = 0; i < 8; ++i) s = fmaxf(s, red[i]);
    red[0] = s;
  }
  __syncthreads();
  return red[0];
}

// ---------------- self-attention (ALiBi, causal, KV-cache) -----------------
// grid = B*H = 64 blocks, 256 threads. kc/vc layout (T,B,D) fp32.
// Output bf16 (consumed only as GEMM A-side).
__global__ __launch_bounds__(256) void self_attn_kernel(
    const float* __restrict__ q, const float* __restrict__ kc,
    const float* __restrict__ vc, bf16* __restrict__ sa, int step)
{
  __shared__ float sc[T_];
  __shared__ __align__(16) float qs[DH_];
  __shared__ float red[8];
  const int b = blockIdx.x >> 2, h = blockIdx.x & 3;
  const float slope = exp2f(-2.0f * (float)(h + 1));   // 0.25^(h+1)
  qs[threadIdx.x] = q[(size_t)b * D_ + h * DH_ + threadIdx.x];
  __syncthreads();
  const float scale = 0.0625f;                          // 1/sqrt(256)
  for (int t = threadIdx.x; t <= step; t += 256) {
    const float4* kp = (const float4*)(kc + ((size_t)t * B_ + b) * D_ + h * DH_);
    const float4* q4 = (const float4*)qs;
    float dot = 0.0f;
#pragma unroll 8
    for (int d = 0; d < DH_ / 4; ++d) {
      float4 a = q4[d], c = kp[d];
      dot += a.x * c.x + a.y * c.y + a.z * c.z + a.w * c.w;
    }
    const int rel = step - t;
    sc[t] = dot * scale - slope * (float)(rel / PERIOD_);
  }
  __syncthreads();
  float mx = -1e30f;
  for (int t = threadIdx.x; t <= step; t += 256) mx = fmaxf(mx, sc[t]);
  mx = block_max256(mx, red);
  float sum = 0.0f;
  for (int t = threadIdx.x; t <= step; t += 256) {
    float e = __expf(sc[t] - mx);
    sc[t] = e;
    sum += e;
  }
  sum = block_sum256(sum, red);
  const float inv = 1.0f / sum;
  const int d = threadIdx.x;
  float acc = 0.0f;
  for (int t = 0; t <= step; ++t)
    acc += sc[t] * vc[((size_t)t * B_ + b) * D_ + h * DH_ + d];
  sa[(size_t)b * D_ + h * DH_ + d] = (bf16)(acc * inv);
}

// ---------------- cross-attention (8-wide window over memory) --------------
// grid = B*H = 64 blocks, 256 threads. mem_k/mem_v (B,S,D) bf16. Out bf16.
__global__ __launch_bounds__(256) void cross_attn_kernel(
    const float* __restrict__ q2, const bf16* __restrict__ mk,
    const bf16* __restrict__ mv, bf16* __restrict__ ca, int step)
{
  __shared__ float p[8];
  __shared__ __align__(16) float qs[DH_];
  const int b = blockIdx.x >> 2, h = blockIdx.x & 3;
  int s0 = 2 * step - 6; if (s0 < 0) s0 = 0;
  int s1 = 2 * step + 1; if (s1 > S_ - 1) s1 = S_ - 1;
  const int nw = s1 - s0 + 1;
  qs[threadIdx.x] = q2[(size_t)b * D_ + h * DH_ + threadIdx.x];
  __syncthreads();
  if (threadIdx.x < nw) {
    const bf16* kp = mk + ((size_t)b * S_ + s0 + threadIdx.x) * D_ + h * DH_;
    float dot = 0.0f;
    for (int d = 0; d < DH_; ++d) dot += qs[d] * (float)kp[d];
    p[threadIdx.x] = dot * 0.0625f;
  }
  __syncthreads();
  if (threadIdx.x == 0) {
    float mx = -1e30f;
    for (int j = 0; j < nw; ++j) mx = fmaxf(mx, p[j]);
    float s = 0.0f;
    for (int j = 0; j < nw; ++j) { p[j] = __expf(p[j] - mx); s += p[j]; }
    const float inv = 1.0f / s;
    for (int j = 0; j < nw; ++j) p[j] *= inv;
  }
  __syncthreads();
  float acc = 0.0f;
  for (int j = 0; j < nw; ++j)
    acc += p[j] * (float)mv[((size_t)b * S_ + s0 + j) * D_ + h * DH_ + threadIdx.x];
  ca[(size_t)b * D_ + h * DH_ + threadIdx.x] = (bf16)acc;
}

// ---------------- layernorm over D=1024, grid = 16 rows, 256 threads -------
// Emits fp32 (for residual paths) and bf16 (for GEMM A-side) copies.
__global__ __launch_bounds__(256) void ln_kernel(
    const float* __restrict__ in, float* __restrict__ outF,
    bf16* __restrict__ outB,
    const float* __restrict__ g, const float* __restrict__ be)
{
  __shared__ float red[8];
  const int b = blockIdx.x;
  const float* row = in + (size_t)b * D_;
  float x[4]; float s = 0.0f;
#pragma unroll
  for (int j = 0; j < 4; ++j) { x[j] = row[threadIdx.x + 256 * j]; s += x[j]; }
  const float mean = block_sum256(s, red) * (1.0f / (float)D_);
  float vs = 0.0f;
#pragma unroll
  for (int j = 0; j < 4; ++j) { float d = x[j] - mean; vs += d * d; }
  const float var = block_sum256(vs, red) * (1.0f / (float)D_);
  const float inv = rsqrtf(var + 1e-5f);
#pragma unroll
  for (int j = 0; j < 4; ++j) {
    const int d = threadIdx.x + 256 * j;
    const float y = (x[j] - mean) * inv * g[d] + be[d];
    outF[(size_t)b * D_ + d] = y;
    outB[(size_t)b * D_ + d] = (bf16)y;
  }
}

// ---------------- small utility kernels ------------------------------------
__global__ void pe_init_kernel(float* __restrict__ pe) {
  const int idx = blockIdx.x * 256 + threadIdx.x;      // PERIOD_*D_ = 25600
  if (idx >= PERIOD_ * D_) return;
  const int p = idx / D_, d = idx % D_;
  const int j = d >> 1;
  const float div = __expf((float)(2 * j) * (-logf(10000.0f) / (float)D_));
  const float ang = (float)p * div;
  pe[idx] = (d & 1) ? __cosf(ang) : __sinf(ang);
}
// x = emb + pe[step%25]; emits fp32 (residual) + bf16 (GEMM A) copies.
__global__ void add_pe_kernel(const float* __restrict__ emb, const float* __restrict__ pe,
                              float* __restrict__ xF, bf16* __restrict__ xB, int step) {
  const int b = blockIdx.x, p = step % PERIOD_;
  for (int d = threadIdx.x; d < D_; d += 256) {
    const float v = emb[(size_t)b * D_ + d] + pe[(size_t)p * D_ + d];
    xF[(size_t)b * D_ + d] = v;
    xB[(size_t)b * D_ + d] = (bf16)v;
  }
}
__global__ void bcast_bias_kernel(const float* __restrict__ bias, float* __restrict__ emb) {
  const int i = blockIdx.x * 256 + threadIdx.x;        // B_*D_ = 16384
  emb[i] = bias[i & (D_ - 1)];
}
__global__ void cvt_bf16_kernel(const float* __restrict__ in, bf16* __restrict__ out, int n) {
  const int i = blockIdx.x * 256 + threadIdx.x;
  if (i < n) out[i] = (bf16)in[i];
}
__global__ void cvt_bf16_pad_kernel(const float* __restrict__ in, bf16* __restrict__ out,
                                    int rows, int K, int Kp) {
  const int i = blockIdx.x * 256 + threadIdx.x;
  if (i >= rows * Kp) return;
  const int r = i / Kp, k = i % Kp;
  out[i] = (k < K) ? (bf16)in[(size_t)r * K + k] : (bf16)0.0f;
}
__global__ __launch_bounds__(256) void loss_reduce_kernel(
    const float* __restrict__ partial, int n, float* __restrict__ out) {
  __shared__ float red[8];
  float s = 0.0f;
  for (int i = threadIdx.x; i < n; i += 256) s += partial[i];
  s = block_sum256(s, red);
  if (threadIdx.x == 0) out[0] = s * (1.0f / (float)((size_t)B_ * T_ * V_));
}

// ---------------------------------------------------------------------------
extern "C" void kernel_launch(void* const* d_in, const int* in_sizes, int n_in,
                              void* d_out, int out_size, void* d_ws, size_t ws_size,
                              hipStream_t stream)
{
  (void)in_sizes; (void)n_in; (void)out_size; (void)ws_size;
  const float* audio   = (const float*)d_in[0];
  const float* vertice = (const float*)d_in[1];
  const float* W_af  = (const float*)d_in[2];  const float* b_af  = (const float*)d_in[3];
  const float* saWq  = (const float*)d_in[4];  const float* sa_bq = (const float*)d_in[5];
  const float* saWk  = (const float*)d_in[6];  const float* sa_bk = (const float*)d_in[7];
  const float* saWv  = (const float*)d_in[8];  const float* sa_bv = (const float*)d_in[9];
  const float* saWo  = (const float*)d_in[10]; const float* sa_bo = (const float*)d_in[11];
  const float* caWq  = (const float*)d_in[12]; const float* ca_bq = (const float*)d_in[13];
  const float* caWk  = (const float*)d_in[14]; const float* ca_bk = (const float*)d_in[15];
  const float* caWv  = (const float*)d_in[16]; const float* ca_bv = (const float*)d_in[17];
  const float* caWo  = (const float*)d_in[18]; const float* ca_bo = (const float*)d_in[19];
  const float* ln1_g = (const float*)d_in[20]; const float* ln1_b = (const float*)d_in[21];
  const float* ln2_g = (const float*)d_in[22]; const float* ln2_b = (const float*)d_in[23];
  const float* ln3_g = (const float*)d_in[24]; const float* ln3_b = (const float*)d_in[25];
  const float* Wff1  = (const float*)d_in[26]; const float* b_ff1 = (const float*)d_in[27];
  const float* Wff2  = (const float*)d_in[28]; const float* b_ff2 = (const float*)d_in[29];
  const float* Wvm   = (const float*)d_in[30]; const float* b_vm  = (const float*)d_in[31];
  const float* Wvr   = (const float*)d_in[32]; const float* b_vr  = (const float*)d_in[33];

  // ---- workspace carve-out (256B aligned) ----
  char* p = (char*)d_ws;
  auto alloc = [&](size_t nbytes) -> void* {
    void* r = (void*)p; p += (nbytes + 255) & ~(size_t)255; return r;
  };
  float* pe    = (float*)alloc((size_t)PERIOD_ * D_ * 4);
  bf16* wafB   = (bf16*)alloc((size_t)D_ * 768 * 2);
  bf16* saWqB  = (bf16*)alloc((size_t)D_ * D_ * 2);
  bf16* saWkB  = (bf16*)alloc((size_t)D_ * D_ * 2);
  bf16* saWvB  = (bf16*)alloc((size_t)D_ * D_ * 2);
  bf16* saWoB  = (bf16*)alloc((size_t)D_ * D_ * 2);
  bf16* caWqB  = (bf16*)alloc((size_t)D_ * D_ * 2);
  bf16* caWkB  = (bf16*)alloc((size_t)D_ * D_ * 2);
  bf16* caWvB  = (bf16*)alloc((size_t)D_ * D_ * 2);
  bf16* caWoB  = (bf16*)alloc((size_t)D_ * D_ * 2);
  bf16* ff1B   = (bf16*)alloc((size_t)FF_ * D_ * 2);
  bf16* ff2B   = (bf16*)alloc((size_t)D_ * FF_ * 2);
  bf16* vrB    = (bf16*)alloc((size_t)V_ * D_ * 2);
  bf16* vmB    = (bf16*)alloc((size_t)D_ * VP_ * 2);     // K padded 1404->1408
  bf16* hidB   = (bf16*)alloc((size_t)B_ * S_ * D_ * 2);
  bf16* mkB    = (bf16*)alloc((size_t)B_ * S_ * D_ * 2);
  bf16* mvB    = (bf16*)alloc((size_t)B_ * S_ * D_ * 2);
  float* kc    = (float*)alloc((size_t)T_ * B_ * D_ * 4);
  float* vc    = (float*)alloc((size_t)T_ * B_ * D_ * 4);
  float* emb   = (float*)alloc((size_t)B_ * D_ * 4);
  float* xb    = (float*)alloc((size_t)B_ * D_ * 4);     // fp32 (residual)
  bf16*  xbB   = (bf16*)alloc((size_t)B_ * D_ * 2);      // bf16 (GEMM A)
  float* x1    = (float*)alloc((size_t)B_ * D_ * 4);
  bf16*  x1B   = (bf16*)alloc((size_t)B_ * D_ * 2);
  float* x2    = (float*)alloc((size_t)B_ * D_ * 4);
  bf16*  x2B   = (bf16*)alloc((size_t)B_ * D_ * 2);
  float* x3    = (float*)alloc((size_t)B_ * D_ * 4);
  bf16*  x3B   = (bf16*)alloc((size_t)B_ * D_ * 2);
  float* qb    = (float*)alloc((size_t)B_ * D_ * 4);     // q stays fp32 for attn
  bf16*  attbB = (bf16*)alloc((size_t)B_ * D_ * 2);      // attn outputs (GEMM A)
  float* yb    = (float*)alloc((size_t)B_ * D_ * 4);     // pre-LN buffer
  bf16*  hbB   = (bf16*)alloc((size_t)B_ * FF_ * 2);     // relu(ff1) bf16
  bf16*  ovp   = (bf16*)alloc((size_t)B_ * VP_ * 2);     // padded out_v bf16
  float* lossp = (float*)alloc((size_t)T_ * VR_BLOCKS * 4);

  auto cvt = [&](const float* in, bf16* out, size_t n) {
    cvt_bf16_kernel<<<dim3((unsigned)((n + 255) / 256)), dim3(256), 0, stream>>>(in, out, (int)n);
  };

  // ---- init + weight conversion ----
  pe_init_kernel<<<dim3(100), dim3(256), 0, stream>>>(pe);
  cvt(W_af, wafB, (size_t)D_ * 768);
  cvt(saWq, saWqB, (size_t)D_ * D_); cvt(saWk, saWkB, (size_t)D_ * D_);
  cvt(saWv, saWvB, (size_t)D_ * D_); cvt(saWo, saWoB, (size_t)D_ * D_);
  cvt(caWq, caWqB, (size_t)D_ * D_); cvt(caWk, caWkB, (size_t)D_ * D_);
  cvt(caWv, caWvB, (size_t)D_ * D_); cvt(caWo, caWoB, (size_t)D_ * D_);
  cvt(Wff1, ff1B, (size_t)FF_ * D_); cvt(Wff2, ff2B, (size_t)D_ * FF_);
  cvt(Wvr, vrB, (size_t)V_ * D_);
  cvt_bf16_pad_kernel<<<dim3((unsigned)(((size_t)D_ * VP_ + 255) / 256)), dim3(256), 0, stream>>>(
      Wvm, vmB, D_, V_, VP_);

  // ---- prefill: hidden / mem_k / mem_v (big parallel WMMA GEMMs) ----
  const dim3 blk128(128), blk256(256);
  gemm_wmma_kernel<false, false, true><<<dim3(16, (B_ * S_) / 16), blk128, 0, stream>>>(
      audio, wafB, b_af, nullptr, hidB, D_, 768);
  gemm_wmma_kernel<true, false, true><<<dim3(16, (B_ * S_) / 16), blk128, 0, stream>>>(
      hidB, caWkB, ca_bk, nullptr, mkB, D_, D_);
  gemm_wmma_kernel<true, false, true><<<dim3(16, (B_ * S_) / 16), blk128, 0, stream>>>(
      hidB, caWvB, ca_bv, nullptr, mvB, D_, D_);
  bcast_bias_kernel<<<dim3(64), blk256, 0, stream>>>(b_vm, emb);

  // ---- sequential decode (KV-cached; equivalent to reference scan) ----
  for (int i = 0; i < T_; ++i) {
    add_pe_kernel<<<dim3(16), blk256, 0, stream>>>(emb, pe, xb, xbB, i);
    gemm_wmma_kernel<true, false, false><<<dim3(16, 1), blk128, 0, stream>>>(
        xbB, saWqB, sa_bq, nullptr, qb, D_, D_);
    gemm_wmma_kernel<true, false, false><<<dim3(16, 1), blk128, 0, stream>>>(
        xbB, saWkB, sa_bk, nullptr, kc + (size_t)i * B_ * D_, D_, D_);
    gemm_wmma_kernel<true, false, false><<<dim3(16, 1), blk128, 0, stream>>>(
        xbB, saWvB, sa_bv, nullptr, vc + (size_t)i * B_ * D_, D_, D_);
    self_attn_kernel<<<dim3(B_ * H_), blk256, 0, stream>>>(qb, kc, vc, attbB, i);
    gemm_wmma_kernel<true, false, false><<<dim3(16, 1), blk128, 0, stream>>>(
        attbB, saWoB, sa_bo, xb, yb, D_, D_);
    ln_kernel<<<dim3(16), blk256, 0, stream>>>(yb, x1, x1B, ln1_g, ln1_b);
    gemm_wmma_kernel<true, false, false><<<dim3(16, 1), blk128, 0, stream>>>(
        x1B, caWqB, ca_bq, nullptr, qb, D_, D_);
    cross_attn_kernel<<<dim3(B_ * H_), blk256, 0, stream>>>(qb, mkB, mvB, attbB, i);
    gemm_wmma_kernel<true, false, false><<<dim3(16, 1), blk128, 0, stream>>>(
        attbB, caWoB, ca_bo, x1, yb, D_, D_);
    ln_kernel<<<dim3(16), blk256, 0, stream>>>(yb, x2, x2B, ln2_g, ln2_b);
    gemm_wmma_kernel<true, true, true><<<dim3(32, 1), blk128, 0, stream>>>(
        x2B, ff1B, b_ff1, nullptr, hbB, FF_, D_);
    gemm_wmma_kernel<true, false, false><<<dim3(16, 1), blk128, 0, stream>>>(
        hbB, ff2B, b_ff2, x2, yb, D_, FF_);
    ln_kernel<<<dim3(16), blk256, 0, stream>>>(yb, x3, x3B, ln3_g, ln3_b);
    vr_loss_kernel<<<dim3(VR_BLOCKS), blk128, 0, stream>>>(
        x3B, vrB, b_vr, vertice, ovp, lossp, i);
    gemm_wmma_kernel<true, false, false><<<dim3(16, 1), blk128, 0, stream>>>(
        ovp, vmB, b_vm, nullptr, emb, D_, VP_);
  }

  loss_reduce_kernel<<<dim3(1), blk256, 0, stream>>>(lossp, T_ * VR_BLOCKS, (float*)d_out);
}
